// DRL_31980326486656
// MI455X (gfx1250) — matrix operand order repaired and checked by reading, not compiled, not run on hardware
//
#include <hip/hip_runtime.h>
#include <hip/hip_bf16.h>

typedef __bf16 bf16_t;
typedef __attribute__((ext_vector_type(16))) __bf16 v16bf;
typedef __attribute__((ext_vector_type(8)))  float  v8f;

#define B_   128
#define S_   128
#define H_   256
#define DYN_ 192
#define K2_  512
#define G3_  768

// ---------------- fast math (gfx1250 TRANS ops, guarded) ----------------
__device__ __forceinline__ float fexp2(float x){
#if __has_builtin(__builtin_amdgcn_exp2f)
  return __builtin_amdgcn_exp2f(x);
#else
  return exp2f(x);
#endif
}
__device__ __forceinline__ float flog2(float x){
#if __has_builtin(__builtin_amdgcn_logf)
  return __builtin_amdgcn_logf(x);
#else
  return log2f(x);
#endif
}
__device__ __forceinline__ float frcp(float x){
#if __has_builtin(__builtin_amdgcn_rcpf)
  return __builtin_amdgcn_rcpf(x);
#else
  return 1.0f / x;
#endif
}
__device__ __forceinline__ float ftanh(float x){
#if __has_builtin(__builtin_amdgcn_tanhf)
  return __builtin_amdgcn_tanhf(x);           // v_tanh_f32
#else
  float e = fexp2(x * 2.8853900817779268f);   // e^{2x}
  return 1.0f - 2.0f * frcp(e + 1.0f);
#endif
}
__device__ __forceinline__ float fexp(float x){ return fexp2(x * 1.4426950408889634f); }
__device__ __forceinline__ float fsig(float x){ return frcp(1.0f + fexp(-x)); }
__device__ __forceinline__ float bf_lo(unsigned u){ return __uint_as_float(u << 16); }
__device__ __forceinline__ float bf_hi(unsigned u){ return __uint_as_float(u & 0xffff0000u); }
__device__ __forceinline__ unsigned short bfbits(float x){
  return __builtin_bit_cast(unsigned short, (bf16_t)x);
}
__device__ __forceinline__ unsigned bfpack(float lo, float hi){
  return (unsigned)bfbits(lo) | ((unsigned)bfbits(hi) << 16);
}
// A-fragment K permutation within a 32-K block:
// positions 0..15 -> K {0..7,16..23} (lanes 0-15), 16..31 -> K {8..15,24..31} (lanes 16-31)
__device__ __forceinline__ int aperm(int ko){
  return (ko < 8) ? ko : (ko < 16) ? (ko + 8) : (ko < 24) ? (ko - 8) : ko;
}

// ---------------- encoders: static_h f32 + SD in WMMA B-fragment layout ------
// SDf[b][kb][s][32]: 32 consecutive K values of block kb for column s.
// Lane (N=s, half hi) loads 16 contiguous bf16 at offset hi*16 -> one 32B vector.
__global__ void k_prep_enc(const float* __restrict__ sta, const float* __restrict__ dyn,
                           const float* __restrict__ esW, const float* __restrict__ esb,
                           const float* __restrict__ edW, const float* __restrict__ edb,
                           float* __restrict__ static_h, bf16_t* __restrict__ SDf,
                           float* __restrict__ lmask)
{
  const int b = blockIdx.x, t = threadIdx.x;
  const float* s1 = sta + ((size_t)b*3 + 1)*S_;     // static[:,1,:], static[:,2,:]
  auto sdf = [&](int k, int s) -> size_t {
    return (((size_t)b*(K2_/32) + (k >> 5))*S_ + s)*32 + (k & 31);
  };
  for (int idx = t; idx < H_*S_; idx += blockDim.x) {
    int h = idx >> 7, s = idx & 127;
    float v = esW[h*2]*s1[s] + esW[h*2+1]*s1[S_+s] + esb[h];
    static_h[((size_t)b*H_ + h)*S_ + s] = v;
    SDf[sdf(h, s)] = (bf16_t)v;
  }
  const float* db = dyn + (size_t)b*DYN_*S_;
  for (int idx = t; idx < DYN_*S_; idx += blockDim.x) {
    int k = idx >> 7, s = idx & 127;
    float acc = edb[k];
    const float* wr = edW + k*DYN_;
    for (int j = 0; j < DYN_; ++j) acc += wr[j]*db[(size_t)j*S_ + s];
    SDf[sdf(H_ + k, s)] = (bf16_t)acc;
  }
  for (int idx = t; idx < (K2_-H_-DYN_)*S_; idx += blockDim.x) {  // zero K pad
    int k = idx >> 7, s = idx & 127;
    SDf[sdf(H_ + DYN_ + k, s)] = (bf16_t)0.0f;
  }
  for (int s = t; s < S_; s += blockDim.x) {
    float mv = 0.f, ma = 0.f, mb = 0.f;
    for (int j = 0; j < 64; ++j) {
      mv += db[(size_t)j*S_ + s];
      ma += db[(size_t)(64+j)*S_ + s];
      mb += db[(size_t)(128+j)*S_ + s];
    }
    float dm = ma*mb + mv;
    lmask[b*S_ + s] = (dm != 0.0f) ? -__builtin_inff() : 0.0f;
  }
}

// ---------------- weight packs + decoder-projection fold ----------------
// Waf/Wpf: A-fragment swizzled [h][kb][32]; Wa3P/Wp3P/WhhP: pair-packed k-major u32.
__global__ void k_prep_w(const float* __restrict__ attnW, const float* __restrict__ ptrW,
                         const float* __restrict__ Wih,   const float* __restrict__ Whhf,
                         const float* __restrict__ bih,   const float* __restrict__ decW,
                         const float* __restrict__ decb,
                         bf16_t* __restrict__ Waf, bf16_t* __restrict__ Wpf,
                         bf16_t* __restrict__ Wa3, bf16_t* __restrict__ Wp3,
                         bf16_t* __restrict__ Whh, float* __restrict__ M1,
                         float* __restrict__ c1)
{
  const int t0 = blockIdx.x*blockDim.x + threadIdx.x;
  const int st = gridDim.x*blockDim.x;
  for (int i = t0; i < H_*K2_; i += st) {            // A-fragment swizzle
    int h = i >> 9, k = i & 511;
    int kb = k >> 5, p = k & 31;
    size_t o = (size_t)h*K2_ + kb*32 + p;
    Waf[o] = (bf16_t)attnW[h*G3_ + k];
    Wpf[o] = (bf16_t)ptrW [h*G3_ + k];
  }
  unsigned* wa3p = (unsigned*)Wa3;
  unsigned* wp3p = (unsigned*)Wp3;
  for (int i = t0; i < H_*(H_/2); i += st) {         // pair-packed k-major
    int r = i >> 7, i2 = i & 127;
    wa3p[(size_t)i2*H_ + r] = bfpack(attnW[r*G3_ + 512 + 2*i2], attnW[r*G3_ + 512 + 2*i2 + 1]);
    wp3p[(size_t)i2*H_ + r] = bfpack(ptrW [r*G3_ + 512 + 2*i2], ptrW [r*G3_ + 512 + 2*i2 + 1]);
  }
  unsigned* whhp = (unsigned*)Whh;
  for (int i = t0; i < G3_*(H_/2); i += st) {
    int r = i >> 7, i2 = i & 127;
    whhp[(size_t)i2*G3_ + r] = bfpack(Whhf[(size_t)r*H_ + 2*i2], Whhf[(size_t)r*H_ + 2*i2 + 1]);
  }
  for (int r = t0; r < G3_; r += st) {               // M1 = Wih@decW, c1 = Wih@decb + bih
    float a0 = 0.f, a1 = 0.f, ac = 0.f;
    const float* wr = Wih + (size_t)r*H_;
    for (int i = 0; i < H_; ++i) {
      float w = wr[i];
      a0 += w*decW[i*2]; a1 += w*decW[i*2+1]; ac += w*decb[i];
    }
    M1[r] = a0; M1[G3_ + r] = a1; c1[r] = ac + bih[r];
  }
}

// ---------------- hoisted GEMM: A_pre/P_pre via v_wmma_f32_16x16x32_bf16 ----
// Fragments are pre-swizzled: each lane loads one aligned 32B vector per operand.
__global__ void k_gemm_pre(const bf16_t* __restrict__ SDf,
                           const bf16_t* __restrict__ Waf, const bf16_t* __restrict__ Wpf,
                           float* __restrict__ A, float* __restrict__ P)
{
  const int wid  = (blockIdx.x*blockDim.x + threadIdx.x) >> 5;
  const int nw   = (gridDim.x*blockDim.x) >> 5;
  const int lane = threadIdx.x & 31;
  const int l    = lane & 15;
  const int hi   = lane >> 4;
  const int TILES = 2*B_*(H_/16)*(S_/16);            // 32768 tiles
  for (int tile = wid; tile < TILES; tile += nw) {
    int m = tile & 1;
    int r = tile >> 1;
    int b  = r & (B_-1);     r >>= 7;
    int ht = (r & 15) << 4;  r >>= 4;
    int st = r << 4;
    const bf16_t* W = m ? Wpf : Waf;
    float*        O = m ? P   : A;
    v8f c = {};
    const bf16_t* ap = W   + (size_t)(ht + l)*K2_ + hi*16;                  // A row M=ht+l
    const bf16_t* bp = SDf + (((size_t)b*(K2_/32))*S_ + st + l)*32 + hi*16; // B col N=st+l
    #pragma unroll 4
    for (int kb = 0; kb < K2_/32; ++kb) {
      v16bf a  = *(const v16bf*)(ap + (size_t)kb*32);
      v16bf bm = *(const v16bf*)(bp + (size_t)kb*(S_*32));
      c = __builtin_amdgcn_wmma_f32_16x16x32_bf16(false, a, false, bm,
                                                  (short)0, c, false, false);
    }
    // C layout: vgpr j -> (M=j(+8 for hi lanes), N=l); store transposed as [b][s][h]
    float* o = O + ((size_t)b*S_ + st + l)*H_ + ht + (hi ? 8 : 0);
    #pragma unroll
    for (int j = 0; j < 8; ++j) o[j] = c[j];
  }
}

// ---------------- persistent decoder: 1 workgroup per batch element ----------
// A_pre[b] + P_pre[b] cached in LDS (2 x 128KB; CDNA5 WGP has 320KB LDS).
__global__ void __launch_bounds__(256) k_decode(
    const float* __restrict__ sta,   const float* __restrict__ dec0,
    const float* __restrict__ static_h, const float* __restrict__ A,
    const float* __restrict__ Pm,    const bf16_t* __restrict__ Wa3,
    const bf16_t* __restrict__ Wp3,  const bf16_t* __restrict__ Whh,
    const float* __restrict__ M1,    const float* __restrict__ c1g,
    const float* __restrict__ bhhg,  const float* __restrict__ attnv,
    const float* __restrict__ ptrv,  const float* __restrict__ lmaskg,
    float* __restrict__ out)
{
  __shared__ __align__(16) float As[S_*H_];   // 128 KB
  __shared__ __align__(16) float Ps[S_*H_];   // 128 KB
  __shared__ float h_lds[H_], aux[H_], ctx[H_];
  __shared__ float ev[S_], pv[S_], lm[S_];
  __shared__ float m1a[G3_], m1b[G3_], c1s[G3_], bhh[G3_];
  __shared__ float va[H_], vp[H_];
  __shared__ float din[2], red[2];
  __shared__ int amax;

  const int b = blockIdx.x, t = threadIdx.x;
  const int wave = t >> 5, lane = t & 31;

  { // vectorized one-time LDS fill of the per-batch precomputed tensors
    const float4* Ab4 = (const float4*)(A  + (size_t)b*S_*H_);
    const float4* Pb4 = (const float4*)(Pm + (size_t)b*S_*H_);
    float4* As4 = (float4*)As;
    float4* Ps4 = (float4*)Ps;
    for (int i = t; i < S_*H_/4; i += 256) { As4[i] = Ab4[i]; Ps4[i] = Pb4[i]; }
  }
  for (int i = t; i < G3_; i += 256) { m1a[i]=M1[i]; m1b[i]=M1[G3_+i]; c1s[i]=c1g[i]; bhh[i]=bhhg[i]; }
  va[t] = attnv[t]; vp[t] = ptrv[t];
  if (t < S_) lm[t] = lmaskg[b*S_ + t];
  if (t < 2)  din[t] = dec0[b*2 + t];
  h_lds[t] = 0.0f;
  __syncthreads();

  const float* SHb = static_h + (size_t)b*H_*S_;
  const float* s1  = sta + ((size_t)b*3 + 1)*S_;
  const unsigned* whhp = (const unsigned*)Whh;   // [k/2][768] pair-packed
  const unsigned* wa3p = (const unsigned*)Wa3;   // [k/2][256]
  const unsigned* wp3p = (const unsigned*)Wp3;

  // per-h gains for the score passes: h = lane + 32*j is wave-invariant across s
  float vh[8], wph[8];
  #pragma unroll
  for (int j = 0; j < 8; ++j) { vh[j] = va[lane + 32*j]; wph[j] = vp[lane + 32*j]; }

  float d0 = din[0], d1 = din[1];

  for (int step = 0; step < S_; ++step) {
    // ---- GRU: gi from folded 768x2 map; gh from pair-packed bf16 W_hh matvec ----
    float gir = m1a[t]*d0      + m1b[t]*d1      + c1s[t];
    float giz = m1a[H_+t]*d0   + m1b[H_+t]*d1   + c1s[H_+t];
    float gin = m1a[2*H_+t]*d0 + m1b[2*H_+t]*d1 + c1s[2*H_+t];
    float ghr = bhh[t], ghz = bhh[H_+t], ghn = bhh[2*H_+t];
    #pragma unroll 4
    for (int i = 0; i < H_/2; ++i) {
      float h0 = h_lds[2*i], h1 = h_lds[2*i+1];
      const unsigned* row = whhp + (size_t)i*G3_;
      unsigned ur = row[t], uz = row[H_ + t], un = row[2*H_ + t];
      ghr += bf_lo(ur)*h0 + bf_hi(ur)*h1;
      ghz += bf_lo(uz)*h0 + bf_hi(uz)*h1;
      ghn += bf_lo(un)*h0 + bf_hi(un)*h1;
    }
    float rg = fsig(gir + ghr);
    float zg = fsig(giz + ghz);
    float ng = ftanh(gin + rg*ghn);
    float hn = (1.0f - zg)*ng + zg*h_lds[t];
    __syncthreads();
    h_lds[t] = hn;
    __syncthreads();

    // ---- attn_h = Wa3 @ h ----
    { float acc = 0.0f;
      #pragma unroll 4
      for (int i = 0; i < H_/2; ++i) {
        unsigned u = wa3p[(size_t)i*H_ + t];
        acc += bf_lo(u)*h_lds[2*i] + bf_hi(u)*h_lds[2*i+1];
      }
      aux[t] = acc; }
    __syncthreads();

    // ---- attention scores: wave per s-column, lanes stride over h ----
    { float ah[8];
      #pragma unroll
      for (int j = 0; j < 8; ++j) ah[j] = aux[lane + 32*j];
      for (int s = wave; s < S_; s += 8) {
        const float* ar = As + (size_t)s*H_ + lane;
        float acc = 0.0f;
        #pragma unroll
        for (int j = 0; j < 8; ++j) acc += vh[j]*ftanh(ar[32*j] + ah[j]);
        #pragma unroll
        for (int off = 16; off; off >>= 1) acc += __shfl_down(acc, off, 32);
        if (lane == 0) ev[s] = acc;
      } }
    __syncthreads();
    if (t < 32) {                                    // softmax stats (wave0)
      float m = -__builtin_inff();
      for (int s = lane; s < S_; s += 32) m = fmaxf(m, ev[s]);
      #pragma unroll
      for (int off = 16; off; off >>= 1) m = fmaxf(m, __shfl_down(m, off, 32));
      m = __shfl(m, 0, 32);
      float sum = 0.0f;
      for (int s = lane; s < S_; s += 32) { float p = fexp(ev[s]-m); pv[s] = p; sum += p; }
      #pragma unroll
      for (int off = 16; off; off >>= 1) sum += __shfl_down(sum, off, 32);
      if (lane == 0) red[0] = frcp(sum);
    }
    __syncthreads();

    // ---- context[h] = sum_s attn[s] * static_h[b][h][s] ----
    { float inv = red[0];
      const float* sh = SHb + (size_t)t*S_;
      float acc = 0.0f;
      #pragma unroll 4
      for (int s = 0; s < S_; ++s) acc += pv[s]*sh[s];
      ctx[t] = acc*inv; }
    __syncthreads();

    // ---- ptr_h = Wp3 @ context ----
    { float acc = 0.0f;
      #pragma unroll 4
      for (int i = 0; i < H_/2; ++i) {
        unsigned u = wp3p[(size_t)i*H_ + t];
        acc += bf_lo(u)*ctx[2*i] + bf_hi(u)*ctx[2*i+1];
      }
      aux[t] = acc; }
    __syncthreads();

    // ---- pointer logits + mask ----
    { float ah[8];
      #pragma unroll
      for (int j = 0; j < 8; ++j) ah[j] = aux[lane + 32*j];
      for (int s = wave; s < S_; s += 8) {
        const float* pr = Ps + (size_t)s*H_ + lane;
        float acc = 0.0f;
        #pragma unroll
        for (int j = 0; j < 8; ++j) acc += wph[j]*ftanh(pr[32*j] + ah[j]);
        #pragma unroll
        for (int off = 16; off; off >>= 1) acc += __shfl_down(acc, off, 32);
        if (lane == 0) ev[s] = acc + lm[s];
      } }
    if (t == 0) amax = 0x7fffffff;
    __syncthreads();
    if (t < 32) {                                    // masked softmax stats
      float m = -__builtin_inff();
      for (int s = lane; s < S_; s += 32) m = fmaxf(m, ev[s]);
      #pragma unroll
      for (int off = 16; off; off >>= 1) m = fmaxf(m, __shfl_down(m, off, 32));
      m = __shfl(m, 0, 32);
      float sum = 0.0f;
      for (int s = lane; s < S_; s += 32) sum += fexp(ev[s]-m);
      #pragma unroll
      for (int off = 16; off; off >>= 1) sum += __shfl_down(sum, off, 32);
      if (lane == 0) { red[0] = m; red[1] = sum; }
    }
    __syncthreads();
    if (t < S_ && ev[t] == red[0]) atomicMin(&amax, t);   // first argmax (jnp.argmax)
    __syncthreads();
    const int ptr = amax;
    if (t == 0) {
      din[0] = s1[ptr];                 // new_dec = static_part[b,:,ptr]
      din[1] = s1[S_ + ptr];
      out[(size_t)b*S_ + step] = (float)ptr;                     // tour_idx
      out[(size_t)B_*S_ + (size_t)b*S_ + step] = -flog2(red[1])*0.6931471805599453f; // logp
    }
    __syncthreads();
    d0 = din[0]; d1 = din[1];
  }
}

// ------------------------------- host launcher -------------------------------
extern "C" void kernel_launch(void* const* d_in, const int* in_sizes, int n_in,
                              void* d_out, int out_size, void* d_ws, size_t ws_size,
                              hipStream_t stream)
{
  const float* sta   = (const float*)d_in[0];
  const float* dyn   = (const float*)d_in[1];
  const float* dec0  = (const float*)d_in[2];
  const float* esW   = (const float*)d_in[3];
  const float* esb   = (const float*)d_in[4];
  const float* edW   = (const float*)d_in[5];
  const float* edb   = (const float*)d_in[6];
  const float* decW  = (const float*)d_in[7];
  const float* decb  = (const float*)d_in[8];
  const float* Wih   = (const float*)d_in[9];
  const float* Whhf  = (const float*)d_in[10];
  const float* bih   = (const float*)d_in[11];
  const float* bhh   = (const float*)d_in[12];
  const float* attnv = (const float*)d_in[13];
  const float* attnW = (const float*)d_in[14];
  const float* ptrv  = (const float*)d_in[15];
  const float* ptrW  = (const float*)d_in[16];

  char* w = (char*)d_ws;
  auto alloc = [&](size_t bytes) -> char* {
    char* p = w; w += (bytes + 255) & ~(size_t)255; return p;
  };
  float*  static_h = (float*) alloc((size_t)B_*H_*S_*4);
  bf16_t* SDf      = (bf16_t*)alloc((size_t)B_*K2_*S_*2);
  float*  A        = (float*) alloc((size_t)B_*S_*H_*4);
  float*  P        = (float*) alloc((size_t)B_*S_*H_*4);
  bf16_t* Waf      = (bf16_t*)alloc((size_t)H_*K2_*2);
  bf16_t* Wpf      = (bf16_t*)alloc((size_t)H_*K2_*2);
  bf16_t* Wa3      = (bf16_t*)alloc((size_t)H_*H_*2);
  bf16_t* Wp3      = (bf16_t*)alloc((size_t)H_*H_*2);
  bf16_t* Whh      = (bf16_t*)alloc((size_t)G3_*H_*2);
  float*  M1       = (float*) alloc((size_t)G3_*2*4);
  float*  c1       = (float*) alloc((size_t)G3_*4);
  float*  lmask    = (float*) alloc((size_t)B_*S_*4);

  k_prep_enc<<<B_, 256, 0, stream>>>(sta, dyn, esW, esb, edW, edb, static_h, SDf, lmask);
  k_prep_w  <<<64, 256, 0, stream>>>(attnW, ptrW, Wih, Whhf, bih, decW, decb,
                                     Waf, Wpf, Wa3, Wp3, Whh, M1, c1);
  k_gemm_pre<<<512, 256, 0, stream>>>(SDf, Waf, Wpf, A, P);
  k_decode  <<<B_, 256, 0, stream>>>(sta, dec0, static_h, A, P, Wa3, Wp3, Whh,
                                     M1, c1, bhh, attnv, ptrv, lmask, (float*)d_out);
}
